// GapLoss_23149873725686
// MI455X (gfx1250) — compile-verified, matrix operand order checked by reading, not run on hardware
//
#include <hip/hip_runtime.h>
#include <cstdint>

// ---------------------------------------------------------------------------
// GapLoss for MI455X (gfx1250, wave32, WGP, 320KB LDS, WMMA)
//
// Pipeline (all on `stream`):
//   K0 binarize : pred -> packed bitmap A (ballot_w32), 8x32768 words in ws
//   K1 skeleton : per-image Zhang-Suen fixpoint, whole image double-buffered
//                 in 256KB of LDS (CDNA5-only), bit-sliced CSA adders,
//                 early-exit on convergence; writes endpoint bitmap C to ws
//   K2 gaploss  : per 16x128 strip: async-stage C halo to LDS, vertical 9-sum
//                 via popcount, horizontal 9-sum via v_wmma_f32_16x16x32_f16,
//                 fused CE(pred,target) * W accumulate, fixed-order reduce
//   K3 finalize : deterministic sum of partials, mean
// ---------------------------------------------------------------------------

#define IMG_H   1024
#define IMG_W   1024
#define NPIX    (IMG_H * IMG_W)       // 1<<20
#define NWORDS  (NPIX / 32)           // 32768 (32-bit words per image)
#define NB      8
#define KGAP    60.0f
#define MAX_ZS_ITERS 256
#define STRIPS_PER_IMG 512            // 64 row-strips(16) x 8 col-strips(128)

typedef __attribute__((ext_vector_type(16))) _Float16 v16h;
typedef __attribute__((ext_vector_type(8)))  float    v8f;

#if __has_builtin(__builtin_amdgcn_global_load_async_to_lds_b32) && \
    __has_builtin(__builtin_amdgcn_s_wait_asynccnt)
#define GAP_ASYNC_LDS 1
typedef __attribute__((address_space(1))) int gas1_int;
typedef __attribute__((address_space(3))) int gas3_int;
#endif

__device__ __forceinline__ unsigned ballot32(bool p) {
#if __has_builtin(__builtin_amdgcn_ballot_w32)
  return __builtin_amdgcn_ballot_w32(p);
#else
  return (unsigned)__ballot((int)p);
#endif
}

// ---------------- bit-sliced adders (32 pixels per op) ----------------------
__device__ __forceinline__ void fadd(unsigned a, unsigned b, unsigned c,
                                     unsigned& s, unsigned& cy) {
  unsigned t = a ^ b;
  s = t ^ c;
  cy = (a & b) | (c & t);
}

// sum of 8 one-bit planes -> 4-bit planes b3..b0 (value 0..8 per pixel)
__device__ __forceinline__ void bitcount8(
    unsigned x0, unsigned x1, unsigned x2, unsigned x3,
    unsigned x4, unsigned x5, unsigned x6, unsigned x7,
    unsigned& b0, unsigned& b1, unsigned& b2, unsigned& b3) {
  unsigned s0, c0, s1, c1, s2, c2, c3, s4, c4, c5;
  fadd(x0, x1, x2, s0, c0);
  fadd(x3, x4, x5, s1, c1);
  fadd(s0, s1, x6, s2, c2);
  b0 = s2 ^ x7;  c3 = s2 & x7;
  fadd(c0, c1, c2, s4, c4);
  b1 = s4 ^ c3;  c5 = s4 & c3;
  b2 = c4 ^ c5;
  b3 = c4 & c5;
}

// Gather the 8 Zhang-Suen neighbor bitmasks for word w (bit i = pixel col c*32+i).
// Zero-padded image borders fall out of the bounds checks.
__device__ __forceinline__ void gather_neighbors(
    const unsigned* src, int w, unsigned cur,
    unsigned& P2, unsigned& P3, unsigned& P4, unsigned& P5,
    unsigned& P6, unsigned& P7, unsigned& P8, unsigned& P9) {
  int row = w >> 5, c = w & 31;
  unsigned um = 0, uc = 0, up = 0, mm = 0, mp = 0, dm = 0, dc = 0, dp = 0;
  if (row > 0) {
    uc = src[w - 32];
    if (c > 0)  um = src[w - 33];
    if (c < 31) up = src[w - 31];
  }
  if (row < IMG_H - 1) {
    dc = src[w + 32];
    if (c > 0)  dm = src[w + 31];
    if (c < 31) dp = src[w + 33];
  }
  if (c > 0)  mm = src[w - 1];
  if (c < 31) mp = src[w + 1];
  P2 = uc;                              // N
  P3 = (uc  >> 1) | (up << 31);         // NE
  P4 = (cur >> 1) | (mp << 31);         // E
  P5 = (dc  >> 1) | (dp << 31);         // SE
  P6 = dc;                              // S
  P7 = (dc  << 1) | (dm >> 31);         // SW
  P8 = (cur << 1) | (mm >> 31);         // W
  P9 = (uc  << 1) | (um >> 31);         // NW
}

// One Zhang-Suen substep over the whole image (32 words per thread).
__device__ unsigned zs_substep(const unsigned* src, unsigned* dst, int step, int tid) {
  unsigned changed = 0;
#pragma unroll 4
  for (int k = 0; k < NWORDS / 1024; ++k) {
    int w = (k << 10) + tid;
    unsigned cur = src[w];
    unsigned P2, P3, P4, P5, P6, P7, P8, P9;
    gather_neighbors(src, w, cur, P2, P3, P4, P5, P6, P7, P8, P9);

    unsigned b0, b1, b2, b3;
    bitcount8(P2, P3, P4, P5, P6, P7, P8, P9, b0, b1, b2, b3);
    // 2 <= Bsum <= 6  :  >=2 is (b1|b2|b3); exclude 7 (b2&b1&b0) and 8 (b3)
    unsigned condB = (b1 | b2 | b3) & ~(b3 | (b2 & b1 & b0));

    // transitions around the ring (P2,P3),(P3,P4)...(P9,P2); exactly one
    unsigned a0, a1, a2, a3;
    bitcount8(~P2 & P3, ~P3 & P4, ~P4 & P5, ~P5 & P6,
              ~P6 & P7, ~P7 & P8, ~P8 & P9, ~P9 & P2, a0, a1, a2, a3);
    unsigned condA = a0 & ~(a1 | a2 | a3);

    unsigned condS = (step == 0)
        ? (~(P2 & P4 & P6) & ~(P4 & P6 & P8))
        : (~(P2 & P4 & P8) & ~(P2 & P6 & P8));

    unsigned del = cur & condB & condA & condS;
    dst[w] = cur & ~del;
    changed |= del;
  }
  return changed;
}

// ---------------- K0: binarize pred -> packed bitmap ------------------------
__global__ __launch_bounds__(256) void k_binarize(const float* __restrict__ pred,
                                                  unsigned* __restrict__ bits) {
  int tid = threadIdx.x;
  long long p = (long long)blockIdx.x * 256 + tid;   // global pixel id, 0..8M-1
  int b   = (int)(p >> 20);
  int pix = (int)(p & (NPIX - 1));
  const float* base = pred + (size_t)b * (2 * NPIX);
  float p0 = base[pix];
  float p1 = base[NPIX + pix];
  unsigned m = ballot32(p1 > p0);                    // argmax>0 with first-wins tie
  if ((tid & 31) == 0) bits[p >> 5] = m;
}

// ---------------- K1: whole-image Zhang-Suen in LDS -------------------------
// 8 blocks x 1024 threads; 256KB + flag of dynamic LDS per workgroup (CDNA5).
__global__ __launch_bounds__(1024) void k_skeleton(unsigned* __restrict__ bits) {
  extern __shared__ unsigned smem[];
  unsigned* bufA = smem;                   // 32768 words
  unsigned* bufB = smem + NWORDS;          // 32768 words
  unsigned* flag = smem + 2 * NWORDS;
  int tid = threadIdx.x;
  unsigned* gbits = bits + (size_t)blockIdx.x * NWORDS;

  for (int k = 0; k < NWORDS / 1024; ++k)
    bufA[(k << 10) + tid] = gbits[(k << 10) + tid];
  __syncthreads();

  for (int it = 0; it < MAX_ZS_ITERS; ++it) {
    if (tid == 0) *flag = 0u;
    __syncthreads();
    unsigned ch = zs_substep(bufA, bufB, 0, tid);
    __syncthreads();
    ch |= zs_substep(bufB, bufA, 1, tid);
    if (ch) atomicOr(flag, 1u);
    __syncthreads();
    unsigned f = *flag;
    __syncthreads();
    if (!f) break;                         // uniform: converged
  }

  // endpoints: skeleton pixel with exactly one 8-neighbor on the skeleton
  for (int k = 0; k < NWORDS / 1024; ++k) {
    int w = (k << 10) + tid;
    unsigned cur = bufA[w];
    unsigned P2, P3, P4, P5, P6, P7, P8, P9;
    gather_neighbors(bufA, w, cur, P2, P3, P4, P5, P6, P7, P8, P9);
    unsigned b0, b1, b2, b3;
    bitcount8(P2, P3, P4, P5, P6, P7, P8, P9, b0, b1, b2, b3);
    gbits[w] = cur & b0 & ~(b1 | b2 | b3);
  }
}

// ---------------- K2: gap weight (WMMA) + CE, fused -------------------------
// Block = 256 threads = 8 waves; block covers a 16-row x 128-col strip;
// each wave produces one 16x16 tile of N via one v_wmma_f32_16x16x32_f16.
__global__ __launch_bounds__(256) void k_gaploss(const float* __restrict__ pred,
                                                 const int* __restrict__ target,
                                                 const unsigned* __restrict__ cbits,
                                                 float* __restrict__ partial) {
  __shared__ unsigned  Cw[24 * 6];       // C bits: 24 halo rows x 6 words
  __shared__ _Float16  Vh[16 * 144];     // vertical 9-sums, cols strip_x-4..+139
  __shared__ float     red[256];

  int tid = threadIdx.x;
  int bid = blockIdx.x;
  int b   = bid >> 9;
  int rem = bid & 511;
  int strip_y = (rem >> 3) << 4;         // 0,16,...,1008
  int strip_x = (rem & 7) << 7;          // 0,128,...,896
  int w0 = strip_x >> 5;
  const unsigned* img = cbits + (size_t)b * NWORDS;

  // ---- stage C halo words (async global->LDS on CDNA5 when available) ----
  if (tid < 144) {
    int rr = tid / 6, wc = tid - rr * 6;
    int row = strip_y - 4 + rr;
    int wi  = w0 - 1 + wc;
    if (row >= 0 && row < IMG_H && wi >= 0 && wi < 32) {
      const unsigned* gp = img + (row << 5) + wi;
#ifdef GAP_ASYNC_LDS
      __builtin_amdgcn_global_load_async_to_lds_b32(
          (gas1_int*)(uintptr_t)gp, (gas3_int*)(uintptr_t)&Cw[tid], 0, 0);
#else
      Cw[tid] = *gp;
#endif
    } else {
      Cw[tid] = 0u;
    }
  }
#ifdef GAP_ASYNC_LDS
  __builtin_amdgcn_s_wait_asynccnt(0);
#endif
  __syncthreads();

  // ---- vertical 9-sums: one column per thread, 24-bit column mask ----
  if (tid < 144) {
    int g = strip_x - 4 + tid;           // global column
    unsigned colmask = 0;
    if (g >= 0 && g < IMG_W) {
      int rel = (g >> 5) - (w0 - 1);
      int bit = g & 31;
#pragma unroll
      for (int rr = 0; rr < 24; ++rr)
        colmask |= ((Cw[rr * 6 + rel] >> bit) & 1u) << rr;
    }
#pragma unroll
    for (int j = 0; j < 16; ++j)
      Vh[j * 144 + tid] = (_Float16)(float)__popc((colmask >> j) & 0x1FFu);
  }
  __syncthreads();

  // ---- horizontal 9-sum via WMMA: D(16x16) = Vext(16x32) x Band(32x16) ----
  int lane   = tid & 31;
  int wid    = tid >> 5;                 // wave -> tile
  int x0     = strip_x + (wid << 4);
  int laneHi = lane >> 4;                // 0 or 1
  int n      = lane & 15;                // A: row M; B: col N

  union { v16h v; _Float16 e[16]; } Af, Bf;
#pragma unroll
  for (int j = 0; j < 8; ++j) {
#pragma unroll
    for (int s = 0; s < 2; ++s) {
      // documented 16-bit A-matrix 16x32 lane/VGPR layout
      int k = ((j >= 4) ? 16 : 0) + (laneHi ? 8 : 0) + ((j & 3) << 1) + s;
      Af.e[2 * j + s] = (k <= 23) ? Vh[n * 144 + (wid << 4) + k]
                                  : (_Float16)0.0f;
      // band: B[k][n] = 1 iff n <= k <= n+8  (K rows 24..31 are zero)
      Bf.e[2 * j + s] = ((k <= 23) && (k >= n) && (k <= n + 8))
                            ? (_Float16)1.0f : (_Float16)0.0f;
    }
  }
  v8f acc = {};
  acc = __builtin_amdgcn_wmma_f32_16x16x32_f16(
      /*neg_a=*/false, Af.v, /*neg_b=*/false, Bf.v,
      /*c_mod=*/(short)0, acc, /*reuse_a=*/false, /*reuse_b=*/false);

  // ---- fused CE * W accumulate (C/D layout: vgpr r -> M=r+8*laneHi, N=n) ----
  union { v8f v; float e[8]; } Nc; Nc.v = acc;
  const float* pb0 = pred + (size_t)b * (2 * NPIX);
  const float* pb1 = pb0 + NPIX;
  const int*   tb  = target + (size_t)b * NPIX;
  __builtin_prefetch(&pb0[((strip_y + (laneHi << 3)) << 10) + x0 + n], 0, 0);
  __builtin_prefetch(&pb1[((strip_y + (laneHi << 3)) << 10) + x0 + n], 0, 0);

  float s = 0.0f;
#pragma unroll
  for (int r = 0; r < 8; ++r) {
    int y = strip_y + (laneHi << 3) + r;
    int x = x0 + n;
    int pix = (y << 10) + x;
    float cnt = Nc.e[r];                        // exact small integer 0..81
    float wgt = (cnt > 0.5f) ? (KGAP * cnt) : 1.0f;
    float p0 = pb0[pix], p1 = pb1[pix];
    float mx = fmaxf(p0, p1);
    float lse = mx + __logf(__expf(p0 - mx) + __expf(p1 - mx));
    float Lce = lse - ((tb[pix] != 0) ? p1 : p0);
    s += wgt * Lce;
  }

  // ---- fixed-order block reduction ----
  red[tid] = s;
  __syncthreads();
#pragma unroll
  for (int off = 128; off > 0; off >>= 1) {
    if (tid < off) red[tid] += red[tid + off];
    __syncthreads();
  }
  if (tid == 0) partial[bid] = red[0];
}

// ---------------- K3: deterministic finalize --------------------------------
__global__ __launch_bounds__(256) void k_finalize(const float* __restrict__ partial,
                                                  int n, float* __restrict__ out) {
  __shared__ float red[256];
  float s = 0.0f;
  for (int i = threadIdx.x; i < n; i += 256) s += partial[i];
  red[threadIdx.x] = s;
  __syncthreads();
#pragma unroll
  for (int off = 128; off > 0; off >>= 1) {
    if (threadIdx.x < off) red[threadIdx.x] += red[threadIdx.x + off];
    __syncthreads();
  }
  if (threadIdx.x == 0) out[0] = red[0] * (1.0f / (float)(NB * NPIX));
}

// ---------------------------------------------------------------------------
extern "C" void kernel_launch(void* const* d_in, const int* in_sizes, int n_in,
                              void* d_out, int out_size, void* d_ws, size_t ws_size,
                              hipStream_t stream) {
  (void)in_sizes; (void)n_in; (void)out_size; (void)ws_size;
  const float* pred   = (const float*)d_in[0];   // [8,2,1024,1024] f32
  const int*   target = (const int*)d_in[1];     // [8,1,1024,1024] int

  unsigned* bits    = (unsigned*)d_ws;                                   // 1 MiB
  float*    partial = (float*)((char*)d_ws + (size_t)NB * NWORDS * 4);   // 16 KiB
  float*    out     = (float*)d_out;

  k_binarize<<<NB * NPIX / 256, 256, 0, stream>>>(pred, bits);
  k_skeleton<<<NB, 1024, (2 * NWORDS + 16) * sizeof(unsigned), stream>>>(bits);
  k_gaploss <<<NB * STRIPS_PER_IMG, 256, 0, stream>>>(pred, target, bits, partial);
  k_finalize<<<1, 256, 0, stream>>>(partial, NB * STRIPS_PER_IMG, out);
}
